// RelAttnBlock_5059471475222
// MI455X (gfx1250) — matrix operand order repaired and checked by reading, not compiled, not run on hardware
//
#include <hip/hip_runtime.h>
#include <hip/hip_bf16.h>
#include <stdint.h>

// ---------------------------------------------------------------------------
// RelAttnBlock for MI455X (gfx1250, wave32, WMMA).
// B=8, L=1024, E=512, H=8, D=64. All matmuls via v_wmma_f32_16x16x32_bf16,
// fp32 accumulation, flash-style fused attention (logits never hit HBM).
// Rel-position skew computed as sliding WMMA "G" tiles + lane-shuffle gather.
// LDS staging uses CDNA5 Tensor Data Mover (tensor_load_to_lds, TENSORcnt,
// 6-arg toolchain variant) and async global->LDS loads (ASYNCcnt), with
// synchronous fallbacks. Workspace requirement: ~43 MB.
// ---------------------------------------------------------------------------

typedef __bf16 bf16_t;
typedef __attribute__((ext_vector_type(16))) __bf16 bf16x16;
typedef __attribute__((ext_vector_type(8)))  float  v8f;
typedef __attribute__((ext_vector_type(4)))  unsigned int u32x4;
typedef __attribute__((ext_vector_type(8)))  int i32x8;
typedef __attribute__((ext_vector_type(4)))  int i32x4;
typedef __attribute__((ext_vector_type(4)))  int v4i;

#define Bn 8
#define Ln 1024
#define En 512
#define Hn 8
#define Dn 64

// ---- CDNA5 data-mover availability ----------------------------------------
#if __has_builtin(__builtin_amdgcn_tensor_load_to_lds) && \
    __has_builtin(__builtin_amdgcn_s_wait_tensorcnt)
#define USE_TDM 1
#else
#define USE_TDM 0
#endif

#if __has_builtin(__builtin_amdgcn_global_load_async_to_lds_b128)
#define USE_ASYNC 1
#else
#define USE_ASYNC 0
#endif

// Per-lane 16B global -> LDS copy. ldsOff is a byte offset into the
// workgroup's dynamic LDS (== AS(3) address, since no static LDS is used).
// Builtin signature (probed): (v4i addrspace(1)*, v4i addrspace(3)*, imm, imm)
__device__ inline void stage_b128(uint32_t ldsOff, const void* gSrc,
                                  char* smembase) {
#if USE_ASYNC
  __builtin_amdgcn_global_load_async_to_lds_b128(
      (__attribute__((address_space(1))) v4i*)(uintptr_t)gSrc,
      (__attribute__((address_space(3))) v4i*)(uintptr_t)ldsOff, 0, 0);
#else
  *(uint4*)(smembase + ldsOff) = *(const uint4*)gSrc;
#endif
}

__device__ inline void wait_async0() {
#if USE_ASYNC
#if __has_builtin(__builtin_amdgcn_s_wait_asynccnt)
  __builtin_amdgcn_s_wait_asynccnt(0);
#else
  asm volatile("s_wait_asynccnt 0" ::: "memory");
#endif
#endif
}

#if USE_TDM
// TDM: DMA a 2-D tile (rows x width bf16 elems, row stride in elems) from
// global memory into LDS at byte offset ldsOff. D# layout per ISA ch.8.
// This toolchain exposes the 6-argument builtin (extra zero int32x8 group).
__device__ inline void tdm_load_2d(uint32_t ldsOff, const void* gaddr,
                                   uint32_t width_elems, uint32_t rows,
                                   uint32_t row_stride_elems) {
  uint64_t ga = (uint64_t)(uintptr_t)gaddr;
  u32x4 g0;
  g0[0] = 1u;                                   // count=1, user mode
  g0[1] = ldsOff;                               // lds_addr (bytes)
  g0[2] = (uint32_t)ga;                         // global_addr[31:0]
  g0[3] = (uint32_t)((ga >> 32) & 0x1FFFFFFu)   // global_addr[56:32]
          | (2u << 30);                         // type = 2 ("image")
  i32x8 g1;
  g1[0] = (int)(1u << 16);                      // data_size = 1 -> 2 bytes
  g1[1] = (int)(width_elems << 16);             // tensor_dim0[15:0]
  g1[2] = (int)((width_elems >> 16) | (rows << 16));  // dim0 hi | tensor_dim1 lo
  g1[3] = (int)((rows >> 16) | (width_elems << 16));  // dim1 hi | tile_dim0
  g1[4] = (int)rows;                            // tile_dim1 (tile_dim2 = 0)
  g1[5] = (int)row_stride_elems;                // tensor_dim0_stride[31:0]
  g1[6] = 0;                                    // stride0 hi | stride1 lo
  g1[7] = 0;                                    // stride1 hi
  i32x4 z4 = {0, 0, 0, 0};
  i32x8 z8 = {0, 0, 0, 0, 0, 0, 0, 0};
  __builtin_amdgcn_tensor_load_to_lds(g0, g1, z4, z4, z8, 0);
}
#endif

// ---- WMMA fragment helpers (wave32, contiguous k = 16*(lane/16) + j in A and B) -------------------

__device__ inline v8f wmma_bf16(bf16x16 a, bf16x16 b, v8f c) {
  return __builtin_amdgcn_wmma_f32_16x16x32_bf16(
      /*neg_a=*/false, a, /*neg_b=*/false, b,
      /*c_mod=*/(short)0, c, /*reuse_a=*/false, /*reuse_b=*/false);
}

// A-matrix 16x32 (MxK), source row-major [row][k], stride ld.
__device__ inline bf16x16 load_a_frag(const bf16_t* src, int ld, int lane) {
  int r    = lane & 15;
  int koff = (lane < 16) ? 0 : 16;
  const bf16_t* p0 = src + r * ld + koff;
  bf16x16 a;
#pragma unroll
  for (int j = 0; j < 16; ++j) a[j] = p0[j];
  return a;
}

// B-matrix 32x16 (KxN), source row-major [n][k] (B[k][n] = src[n][k]).
__device__ inline bf16x16 load_b_frag_nk(const bf16_t* src, int ld, int lane) {
  int n    = lane & 15;
  int koff = (lane < 16) ? 0 : 16;
  const bf16_t* p = src + n * ld + koff;
  bf16x16 b;
#pragma unroll
  for (int j = 0; j < 16; ++j) b[j] = p[j];
  return b;
}

// B-matrix 32x16 (KxN), source row-major [k][n], stride ld.
__device__ inline bf16x16 load_b_frag_kn(const bf16_t* src, int ld, int lane) {
  int n    = lane & 15;
  int koff = (lane < 16) ? 0 : 16;
  bf16x16 b;
#pragma unroll
  for (int j = 0; j < 16; ++j) b[j] = src[(koff + j) * ld + n];
  return b;
}

// ---- elementwise kernels ---------------------------------------------------

__global__ void cvt_bf16_kernel(const float* __restrict__ src,
                                bf16_t* __restrict__ dst, int n) {
  for (int i = blockIdx.x * blockDim.x + threadIdx.x; i < n;
       i += gridDim.x * blockDim.x)
    dst[i] = (bf16_t)src[i];
}

// LayerNorm: one row (512 elems) per 256-thread block -> bf16
__global__ __launch_bounds__(256) void ln_kernel(const float* __restrict__ x,
                                                 const float* __restrict__ gamma,
                                                 const float* __restrict__ beta,
                                                 bf16_t* __restrict__ hbf) {
  int row = blockIdx.x;
  int t   = threadIdx.x;
  const float* xr = x + (size_t)row * En;
  float v0 = xr[t], v1 = xr[t + 256];
  float s = v0 + v1, ss = v0 * v0 + v1 * v1;
#pragma unroll
  for (int off = 16; off > 0; off >>= 1) {
    s  += __shfl_xor(s, off, 32);
    ss += __shfl_xor(ss, off, 32);
  }
  __shared__ float sb[8], ssb[8];
  int w = t >> 5, lane = t & 31;
  if (lane == 0) { sb[w] = s; ssb[w] = ss; }
  __syncthreads();
  float ts = 0.f, tss = 0.f;
#pragma unroll
  for (int i = 0; i < 8; ++i) { ts += sb[i]; tss += ssb[i]; }
  float mean = ts * (1.0f / En);
  float var  = tss * (1.0f / En) - mean * mean;
  float inv  = rsqrtf(var + 1e-5f);
  bf16_t* hr = hbf + (size_t)row * En;
  hr[t]       = (bf16_t)((v0 - mean) * inv * gamma[t] + beta[t]);
  hr[t + 256] = (bf16_t)((v1 - mean) * inv * gamma[t + 256] + beta[t + 256]);
}

// ---- generic GEMM: C = A(MxK) * W^T(NxK) + bias ----------------------------
// 128x128 block tile, 8 waves (4x2), each wave 32x64 = 2x4 WMMA tiles.
// A/W k-chunks staged into LDS by the Tensor Data Mover (wave 0 issues the
// descriptors) or async/sync per-lane copies.
// mode 0: fp32 row-major out; mode 1: bf16 head-scatter (q/k/v).
__global__ __launch_bounds__(256) void gemm_kernel(const bf16_t* __restrict__ A,
                                                   const bf16_t* __restrict__ W,
                                                   const float* __restrict__ bias,
                                                   float* __restrict__ outF,
                                                   bf16_t* __restrict__ outB,
                                                   int M, int N, int K, int mode) {
  extern __shared__ char smem[];
  bf16_t* As = (bf16_t*)smem;        // [128][64]  byte off 0
  bf16_t* Ws = As + 128 * 64;        // [128][64]  byte off 16384

  int m0 = blockIdx.y * 128, n0 = blockIdx.x * 128;
  int t = threadIdx.x, w = t >> 5, lane = t & 31;
  int waveM = w >> 1, waveN = w & 1;

  v8f acc[2][4];
#pragma unroll
  for (int mt = 0; mt < 2; ++mt)
#pragma unroll
    for (int nt = 0; nt < 4; ++nt) acc[mt][nt] = {};

  for (int k0 = 0; k0 < K; k0 += 64) {
    __syncthreads();
#if USE_TDM
    if (w == 0) {
      tdm_load_2d(0u,     A + (size_t)m0 * K + k0, 64u, 128u, (uint32_t)K);
      tdm_load_2d(16384u, W + (size_t)n0 * K + k0, 64u, 128u, (uint32_t)K);
    }
#else
#pragma unroll
    for (int it = 0; it < 4; ++it) {
      int slot = t + 256 * it;
      int row = slot >> 3, grp = slot & 7;
      stage_b128((uint32_t)(row * 128 + grp * 16),
                 A + (size_t)(m0 + row) * K + k0 + grp * 8, smem);
      stage_b128((uint32_t)(16384 + row * 128 + grp * 16),
                 W + (size_t)(n0 + row) * K + k0 + grp * 8, smem);
    }
#endif
    // prefetch next k-chunk while DMA/WMMA work proceeds
    if (k0 + 64 < K) {
      const char* an = (const char*)(A + (size_t)(m0 + (t >> 1)) * K + k0 + 64);
      const char* wn = (const char*)(W + (size_t)(n0 + (t >> 1)) * K + k0 + 64);
      __builtin_prefetch(an + (t & 1) * 64, 0, 1);
      __builtin_prefetch(wn + (t & 1) * 64, 0, 1);
    }
    wait_async0();
#if USE_TDM
    if (w == 0) __builtin_amdgcn_s_wait_tensorcnt(0);
#endif
    __syncthreads();
#pragma unroll
    for (int ks = 0; ks < 64; ks += 32) {
      bf16x16 af[2], wf[4];
#pragma unroll
      for (int mt = 0; mt < 2; ++mt)
        af[mt] = load_a_frag(As + (32 * waveM + 16 * mt) * 64 + ks, 64, lane);
#pragma unroll
      for (int nt = 0; nt < 4; ++nt)
        wf[nt] = load_b_frag_nk(Ws + (64 * waveN + 16 * nt) * 64 + ks, 64, lane);
#pragma unroll
      for (int mt = 0; mt < 2; ++mt)
#pragma unroll
        for (int nt = 0; nt < 4; ++nt)
          acc[mt][nt] = wmma_bf16(af[mt], wf[nt], acc[mt][nt]);
    }
  }

  int scol = lane & 15, upper = (lane >> 4) & 1;
#pragma unroll
  for (int mt = 0; mt < 2; ++mt)
#pragma unroll
    for (int nt = 0; nt < 4; ++nt)
#pragma unroll
      for (int i = 0; i < 8; ++i) {
        int r = m0 + 32 * waveM + 16 * mt + i + upper * 8;
        int c = n0 + 64 * waveN + 16 * nt + scol;
        float v = acc[mt][nt][i] + bias[c];
        if (mode == 0) {
          outF[(size_t)r * N + c] = v;
        } else {
          int bb = r >> 10, l = r & 1023, hh = c >> 6, dd = c & 63;
          outB[((size_t)((bb * Hn + hh) * Ln + l)) * Dn + dd] = (bf16_t)v;
        }
      }
}

// ---- fused causal attention with relative positions ------------------------
// grid = (L/128 row-blocks, B*H). 8 waves, wave w owns rows [l0+16w, l0+16w+16).
// SRel[l,c] = q[l] . Er[1023-l+c] (c<=l): per 16x16 tile the Er index is
// base + s - r, covered by two adjacent WMMA G tiles over a 256-row Er window
// in LDS; diagonal extracted with lane shuffles (C layout: col = lane).
// K/V tiles staged by TDM; Er window rows (clamped gather) by async loads.
#define KT_OFF 0u
#define VT_OFF 16384u
#define ER_OFF 32768u
#define PB_OFF 65536u

__global__ __launch_bounds__(256) void attn_kernel(const bf16_t* __restrict__ qbf,
                                                   const bf16_t* __restrict__ kbf,
                                                   const bf16_t* __restrict__ vbf,
                                                   const bf16_t* __restrict__ erbf,
                                                   bf16_t* __restrict__ attbf) {
  extern __shared__ char smem[];
  bf16_t* Ktile = (bf16_t*)(smem + KT_OFF);   // [128][64]   16 KB
  bf16_t* Vtile = (bf16_t*)(smem + VT_OFF);   // [128][64]   16 KB
  bf16_t* ErW   = (bf16_t*)(smem + ER_OFF);   // [256][64]   32 KB
  bf16_t* Pbuf  = (bf16_t*)(smem + PB_OFF);   // [128][128]  32 KB

  int rb = blockIdx.x;
  int bh = blockIdx.y;          // b*H + h
  int b  = bh >> 3, h = bh & 7;
  int l0 = rb * 128;
  int t = threadIdx.x, w = t >> 5, lane = t & 31;
  int scol = lane & 15, upper = (lane >> 4) & 1;
  const float scale = 0.125f;   // D^-0.5

  // Q rows for this wave stay in registers (two K-halves of 16x64)
  const bf16_t* qrow = qbf + ((size_t)(bh * Ln + l0 + 16 * w)) * Dn;
  bf16x16 aq0 = load_a_frag(qrow, Dn, lane);
  bf16x16 aq1 = load_a_frag(qrow + 32, Dn, lane);

  float mrow[8], lrow[8];
  v8f o[4];
#pragma unroll
  for (int i = 0; i < 8; ++i) { mrow[i] = -1e30f; lrow[i] = 0.f; }
#pragma unroll
  for (int dt = 0; dt < 4; ++dt) o[dt] = {};

  for (int cb = 0; cb <= rb; ++cb) {
    int c0 = cb * 128;
    __syncthreads();
    const bf16_t* kg = kbf + ((size_t)(bh * Ln + c0)) * Dn;
    const bf16_t* vg = vbf + ((size_t)(bh * Ln + c0)) * Dn;
#if USE_TDM
    if (w == 0) {
      tdm_load_2d(KT_OFF, kg, 64u, 128u, 64u);
      tdm_load_2d(VT_OFF, vg, 64u, 128u, 64u);
    }
#else
#pragma unroll
    for (int it = 0; it < 4; ++it) {
      int idx = t + 256 * it;
      stage_b128(KT_OFF + idx * 16, (const char*)kg + idx * 16, smem);
      stage_b128(VT_OFF + idx * 16, (const char*)vg + idx * 16, smem);
    }
#endif
    {
      int jwin0 = 895 - l0 + c0;  // covers all j = base + s - r used below
#pragma unroll
      for (int it = 0; it < 8; ++it) {
        int slot = t + 256 * it;
        int row = slot >> 3, grp = slot & 7;
        int j = jwin0 + row;
        j = j < 0 ? 0 : (j > 1023 ? 1023 : j);  // clamped rows feed masked elems only
        stage_b128(ER_OFF + (uint32_t)(row * 128 + grp * 16),
                   erbf + ((size_t)(h * Ln + j)) * Dn + grp * 8, smem);
      }
    }
    // prefetch next column block's K/V while this one is consumed
    if (cb < rb) {
      const char* kn = (const char*)(kg + (size_t)128 * Dn);
      const char* vn = (const char*)(vg + (size_t)128 * Dn);
      __builtin_prefetch(kn + t * 64, 0, 1);
      __builtin_prefetch(vn + t * 64, 0, 1);
    }
    wait_async0();
#if USE_TDM
    if (w == 0) __builtin_amdgcn_s_wait_tensorcnt(0);
#endif
    __syncthreads();

    // ---- logits: S = scale * Q.K^T + SRel, causal mask -------------------
    float sv[8][8];  // [ct][elem]
    int wb0 = 112 - 16 * w;  // window row of the first "previous" G tile
    v8f gprev = {};
    gprev = wmma_bf16(aq0, load_b_frag_nk(ErW + wb0 * 64, 64, lane), gprev);
    gprev = wmma_bf16(aq1, load_b_frag_nk(ErW + wb0 * 64 + 32, 64, lane), gprev);
#pragma unroll
    for (int ct = 0; ct < 8; ++ct) {
      int wbc = 128 - 16 * w + 16 * ct;
      v8f gcur = {};
      gcur = wmma_bf16(aq0, load_b_frag_nk(ErW + wbc * 64, 64, lane), gcur);
      gcur = wmma_bf16(aq1, load_b_frag_nk(ErW + wbc * 64 + 32, 64, lane), gcur);
      v8f s = {};
      s = wmma_bf16(aq0, load_b_frag_nk(Ktile + 16 * ct * 64, 64, lane), s);
      s = wmma_bf16(aq1, load_b_frag_nk(Ktile + 16 * ct * 64 + 32, 64, lane), s);
#pragma unroll
      for (int i = 0; i < 8; ++i) {
        int r = i + (upper ? 8 : 0);
        int srcLane = (lane & 16) | ((scol - r) & 15);
        float g0 = __shfl(gprev[i], srcLane, 32);
        float g1 = __shfl(gcur[i], srcLane, 32);
        float rel = (scol >= r) ? g1 : g0;   // diagonal gather: col = s - r
        int gl = l0 + 16 * w + r;
        int gc = c0 + 16 * ct + scol;
        float logit = s[i] * scale + rel;
        sv[ct][i] = (gc > gl) ? -1e30f : logit;
      }
      gprev = gcur;
    }

    // ---- online softmax over the whole 128-col block ---------------------
    float bm[8];
#pragma unroll
    for (int i = 0; i < 8; ++i) {
      float m = sv[0][i];
#pragma unroll
      for (int ct = 1; ct < 8; ++ct) m = fmaxf(m, sv[ct][i]);
#pragma unroll
      for (int off = 1; off <= 8; off <<= 1) m = fmaxf(m, __shfl_xor(m, off, 32));
      bm[i] = m;
    }
    float alpha[8];
#pragma unroll
    for (int i = 0; i < 8; ++i) {
      float mnew = fmaxf(mrow[i], bm[i]);
      alpha[i]   = __expf(mrow[i] - mnew);
      mrow[i]    = mnew;
      lrow[i] *= alpha[i];
    }
#pragma unroll
    for (int dt = 0; dt < 4; ++dt)
#pragma unroll
      for (int i = 0; i < 8; ++i) o[dt][i] *= alpha[i];

    // P = exp(S - m) -> bf16 LDS strip (per-lane partial row sums in lrow)
#pragma unroll
    for (int ct = 0; ct < 8; ++ct)
#pragma unroll
      for (int i = 0; i < 8; ++i) {
        float p = __expf(sv[ct][i] - mrow[i]);
        lrow[i] += p;
        int r = i + (upper ? 8 : 0);
        Pbuf[(16 * w + r) * 128 + 16 * ct + scol] = (bf16_t)p;
      }
    __syncthreads();

    // ---- O += P.V --------------------------------------------------------
#pragma unroll
    for (int kk = 0; kk < 4; ++kk) {
      bf16x16 ap = load_a_frag(Pbuf + (16 * w) * 128 + 32 * kk, 128, lane);
#pragma unroll
      for (int dt = 0; dt < 4; ++dt) {
        bf16x16 bv_ = load_b_frag_kn(Vtile + (32 * kk) * 64 + 16 * dt, 64, lane);
        o[dt] = wmma_bf16(ap, bv_, o[dt]);
      }
    }
  }

  // ---- normalize and store [b][l][h*64+d] (bf16, feeds output projection) --
  float invr[8];
#pragma unroll
  for (int i = 0; i < 8; ++i) {
    float rs = lrow[i];
#pragma unroll
    for (int off = 1; off <= 8; off <<= 1) rs += __shfl_xor(rs, off, 32);
    invr[i] = 1.0f / rs;
  }
#pragma unroll
  for (int dt = 0; dt < 4; ++dt)
#pragma unroll
    for (int i = 0; i < 8; ++i) {
      int r = i + (upper ? 8 : 0);
      int l = l0 + 16 * w + r;
      size_t idx = ((size_t)(b * Ln + l)) * En + h * Dn + 16 * dt + scol;
      attbf[idx] = (bf16_t)(o[dt][i] * invr[i]);
    }
}

// ---------------------------------------------------------------------------

extern "C" void kernel_launch(void* const* d_in, const int* in_sizes, int n_in,
                              void* d_out, int out_size, void* d_ws, size_t ws_size,
                              hipStream_t stream) {
  (void)in_sizes; (void)n_in; (void)out_size; (void)ws_size;
  const float* x     = (const float*)d_in[0];
  /* d_in[1] = mask (bool) — causality applied analytically */
  const float* gamma = (const float*)d_in[2];
  const float* beta  = (const float*)d_in[3];
  const float* Wq = (const float*)d_in[4];  const float* bq = (const float*)d_in[5];
  const float* Wk = (const float*)d_in[6];  const float* bk = (const float*)d_in[7];
  const float* Wv = (const float*)d_in[8];  const float* bv = (const float*)d_in[9];
  const float* Wo = (const float*)d_in[10]; const float* bo = (const float*)d_in[11];
  const float* Er = (const float*)d_in[12];
  float* out = (float*)d_out;

  char* ws = (char*)d_ws;
  size_t off = 0;
  auto alloc = [&](size_t bytes) -> void* {
    void* p = ws + off;
    off = (off + bytes + 255) & ~(size_t)255;
    return p;
  };
  const size_t ML = (size_t)Bn * Ln;                     // 8192
  bf16_t* hbf   = (bf16_t*)alloc(ML * En * 2);           // LN output
  bf16_t* wqbf  = (bf16_t*)alloc((size_t)En * En * 2);
  bf16_t* wkbf  = (bf16_t*)alloc((size_t)En * En * 2);
  bf16_t* wvbf  = (bf16_t*)alloc((size_t)En * En * 2);
  bf16_t* wobf  = (bf16_t*)alloc((size_t)En * En * 2);
  bf16_t* erbf  = (bf16_t*)alloc((size_t)Hn * Ln * Dn * 2);
  bf16_t* qbf   = (bf16_t*)alloc((size_t)Bn * Hn * Ln * Dn * 2);
  bf16_t* kbf   = (bf16_t*)alloc((size_t)Bn * Hn * Ln * Dn * 2);
  bf16_t* vbf   = (bf16_t*)alloc((size_t)Bn * Hn * Ln * Dn * 2);
  bf16_t* attbf = (bf16_t*)alloc(ML * En * 2);
  // total workspace ~43 MB

  // 1) LayerNorm -> bf16
  ln_kernel<<<(int)ML, 256, 0, stream>>>(x, gamma, beta, hbf);

  // 2) weight / Er conversion to bf16
  cvt_bf16_kernel<<<256, 256, 0, stream>>>(Wq, wqbf, En * En);
  cvt_bf16_kernel<<<256, 256, 0, stream>>>(Wk, wkbf, En * En);
  cvt_bf16_kernel<<<256, 256, 0, stream>>>(Wv, wvbf, En * En);
  cvt_bf16_kernel<<<256, 256, 0, stream>>>(Wo, wobf, En * En);
  cvt_bf16_kernel<<<256, 256, 0, stream>>>(Er, erbf, Hn * Ln * Dn);

  // 3) Q/K/V projections (WMMA GEMM, scatter to [b][h][l][d] bf16)
  dim3 gg(En / 128, (unsigned)(ML / 128));
  gemm_kernel<<<gg, 256, 32768, stream>>>(hbf, wqbf, bq, nullptr, qbf,
                                          (int)ML, En, En, 1);
  gemm_kernel<<<gg, 256, 32768, stream>>>(hbf, wkbf, bk, nullptr, kbf,
                                          (int)ML, En, En, 1);
  gemm_kernel<<<gg, 256, 32768, stream>>>(hbf, wvbf, bv, nullptr, vbf,
                                          (int)ML, En, En, 1);

  // 4) fused causal rel-position attention (flash-style, all WMMA)
  attn_kernel<<<dim3(Ln / 128, Bn * Hn), 256, 98304, stream>>>(qbf, kbf, vbf,
                                                               erbf, attbf);

  // 5) output projection -> fp32 out
  gemm_kernel<<<gg, 256, 32768, stream>>>(attbf, wobf, bo, out, nullptr,
                                          (int)ML, En, En, 0);
}
